// TemporalAttention_24369644438027
// MI455X (gfx1250) — matrix operand order repaired and checked by reading, not compiled, not run on hardware
//
#include <hip/hip_runtime.h>

// ---------------- problem dimensions ----------------
#define BATCH  16384
#define NBR    20
#define NODE_D 172
#define EDGE_D 172
#define TIME_D 100
#define OUT_D  272          // NODE_D + TIME_D
#define KEY_D  444          // NODE_D + EDGE_D + TIME_D
#define OUT_P  288          // OUT_D padded to 9*32
#define KEY_P  448          // KEY_D padded to 14*32
#define HD     136          // OUT_D / 2 heads
#define HDP    160          // HD padded to 5*32
#define TB     16           // batch elements per workgroup
#define LN_EPS 1e-5f
#define INV_SQRT_HD 0.08574929257125441f   // 136^-0.5

// ---------------- fragment grids (B-matrices prepacked in ws) ----------------
#define QF_KT 9
#define QF_NT 17
#define KF_KT 5
#define KF_NT 28
#define VF_KT 14
#define VF_NT 9
#define OF_KT 9
#define OF_NT 17

#define NFRAG_Q (QF_KT*QF_NT)          // 153
#define NFRAG_K (2*KF_KT*KF_NT)        // 280
#define NFRAG_V (2*VF_KT*VF_NT)        // 252
#define NFRAG_O (OF_KT*OF_NT)          // 153
#define OFF_WQF 0
#define OFF_WKF (NFRAG_Q*512)                  // 78336
#define OFF_WVF (OFF_WKF + NFRAG_K*512)        // 221696
#define OFF_WOF (OFF_WVF + NFRAG_V*512)        // 350720
#define WS_HALVES (OFF_WOF + NFRAG_O*512)      // 429056 halves = 858112 B

// ---------------- LDS layout (manual placement, lifetime reuse) --------------
#define SM_QA 0                        // 16x288 bf16 q tile (A layout src)    9216
#define SM_TQ 9216                     // 2x16x160 bf16 per-head Q (A src)    10240
#define SM_T  19456                    // 2x16x448 bf16 T matrix; later f32 xbuf (17408) 28672
#define SM_KV 48128                    // 2 x 35520 f32 kv double buffer; later oA bf16 (9216) 71040
#define SM_CA 119168                   // 2x16x448 bf16 aggregated c (A src)  28672
#define SM_SC 147840                   // scores (40 f32) + attn weights (40 f32)  320
#define SM_TOTAL 148160
#define KV_FLOATS (NBR*KEY_D)          // 8880
#define KV_BYTES  (KV_FLOATS*4)        // 35520

typedef __attribute__((ext_vector_type(16))) __bf16 v16bf;
typedef __attribute__((ext_vector_type(8)))  __bf16 v8bf;
typedef __attribute__((ext_vector_type(8)))  float  v8f;
typedef __attribute__((ext_vector_type(4)))  unsigned int u32x4;
typedef __attribute__((ext_vector_type(8)))  int i32x8;
typedef __attribute__((ext_vector_type(4)))  int i32x4;

#if __has_builtin(__builtin_amdgcn_tensor_load_to_lds)
#define HAVE_TDM 1
#else
#define HAVE_TDM 0
#endif

// ---------------- WMMA fragment helpers --------------------------------------
// A fragment (16x32 bf16 tile, row-major LDS, strideH in halves).
// ISA 7.12.2: lane<16 holds row=lane, K={0..7,16..23}; lane>=16 holds K={8..15,24..31}.
__device__ __forceinline__ v16bf load_a_frag(const __bf16* base, int strideH, int kt, int lane) {
    int row = lane & 15;
    int kb  = kt * 32 + ((lane & 16) ? 8 : 0);
    const __bf16* p = base + row * strideH + kb;
    v8bf lo = *(const v8bf*)(p);
    v8bf hi = *(const v8bf*)(p + 16);
    return __builtin_shufflevector(lo, hi, 0,1,2,3,4,5,6,7,8,9,10,11,12,13,14,15);
}

// B fragment: prepacked per-lane contiguous (512 halves / fragment, 16 per lane).
__device__ __forceinline__ v16bf load_b_frag(const __bf16* frag, int lane) {
    return *(const v16bf*)(frag + lane * 16);
}

__device__ __forceinline__ v8f wmma_bf16(v16bf a, v16bf b, v8f c) {
    return __builtin_amdgcn_wmma_f32_16x16x32_bf16(false, a, false, b, (short)0, c, false, false);
}

// ---------------- TDM: 2D f32 tile -> LDS ------------------------------------
#if HAVE_TDM
__device__ __forceinline__ void tdm_load_2d(const float* gptr, unsigned ldsoff, int dim0, int dim1) {
    unsigned long long ga = (unsigned long long)gptr;
    u32x4 g0 = { 1u,                                            // count=1, not restore, no gather
                 ldsoff,                                        // lds_addr [63:32]
                 (unsigned)ga,                                  // global_addr[31:0]
                 ((unsigned)(ga >> 32) & 0x01ffffffu) | (2u << 30) }; // global_addr[56:32] | type=2
    i32x8 g1 = { (2 << 16),                                     // data_size=4B, mask=0, no pad
                 (dim0 & 0xffff) << 16,                         // tensor_dim0[15:0]
                 ((int)((unsigned)dim0 >> 16)) | ((dim1 & 0xffff) << 16), // td0 hi | td1 lo
                 ((int)((unsigned)dim1 >> 16)) | ((dim0 & 0xffff) << 16), // td1 hi | tile_dim0
                 (dim1 & 0xffff),                               // tile_dim1, tile_dim2=0
                 dim0,                                          // tensor_dim0_stride lo32
                 0, 0 };
    i32x4 gz = {0, 0, 0, 0};
    i32x8 gz8 = {0, 0, 0, 0, 0, 0, 0, 0};
    // 6-arg toolchain variant: (g0, g1, g2, g3, g4, cpol)
    __builtin_amdgcn_tensor_load_to_lds(g0, g1, gz, gz, gz8, 0);
}

__device__ __forceinline__ void issue_kv(const float* nbrN, const float* nbrE, const float* nbrT,
                                         int bElem, int buf) {
    unsigned base = SM_KV + (unsigned)buf * KV_BYTES;
    tdm_load_2d(nbrN + (size_t)bElem * NBR * NODE_D, base,                        NODE_D, NBR);
    tdm_load_2d(nbrE + (size_t)bElem * NBR * EDGE_D, base + NBR * NODE_D * 4,     EDGE_D, NBR);
    tdm_load_2d(nbrT + (size_t)bElem * NBR * TIME_D, base + 2 * NBR * NODE_D * 4, TIME_D, NBR);
}
#endif

// ---------------- prep kernel: f32 weights -> bf16 B-fragments in ws ---------
__global__ __launch_bounds__(256) void tga_prep(const float* __restrict__ WQ,
                                                const float* __restrict__ WK,
                                                const float* __restrict__ WV,
                                                const float* __restrict__ WO,
                                                unsigned short* __restrict__ wsu) {
    __bf16* ws = (__bf16*)wsu;
    int idx = blockIdx.x * 256 + threadIdx.x;
    if (idx >= WS_HALVES) return;
    int lane = (idx >> 4) & 31;        // section offsets are 512-half aligned
    int i    = idx & 15;
    int kk   = (i & 7) + ((i & 8) << 1) + ((lane & 16) >> 1);   // K within 32-row B tile
    int nn   = lane & 15;                  // N within 16-col B tile
    float v = 0.f;
    if (idx < OFF_WKF) {                                   // W_Q^T tiles: B[k][n]=WQ[n][k]
        int f = idx >> 9;
        int kt = f / QF_NT, nt = f % QF_NT;
        int K = kt * 32 + kk, N = nt * 16 + nn;
        if (K < OUT_D && N < OUT_D) v = WQ[N * OUT_D + K];
    } else if (idx < OFF_WVF) {                            // per-head W_K: B[d][k']=WK[h*HD+d][k']
        int f = (idx - OFF_WKF) >> 9;
        int h = f / (KF_KT * KF_NT); f %= (KF_KT * KF_NT);
        int kt = f / KF_NT, nt = f % KF_NT;
        int D = kt * 32 + kk, K = nt * 16 + nn;
        if (D < HD && K < KEY_D) v = WK[(h * HD + D) * KEY_D + K];
    } else if (idx < OFF_WOF) {                            // per-head W_V^T: B[k'][d]=WV[h*HD+d][k']
        int f = (idx - OFF_WVF) >> 9;
        int h = f / (VF_KT * VF_NT); f %= (VF_KT * VF_NT);
        int kt = f / VF_NT, nt = f % VF_NT;
        int K = kt * 32 + kk, D = nt * 16 + nn;
        if (K < KEY_D && D < HD) v = WV[(h * HD + D) * KEY_D + K];
    } else {                                               // W_O^T tiles
        int f = (idx - OFF_WOF) >> 9;
        int kt = f / OF_NT, nt = f % OF_NT;
        int K = kt * 32 + kk, N = nt * 16 + nn;
        if (K < OUT_D && N < OUT_D) v = WO[N * OUT_D + K];
    }
    ws[idx] = (__bf16)v;
}

// ---------------- fused main kernel ------------------------------------------
__global__ __launch_bounds__(256) void tga_main(const float* __restrict__ node,
                                                const float* __restrict__ timef,
                                                const float* __restrict__ nbrN,
                                                const float* __restrict__ nbrT,
                                                const float* __restrict__ nbrE,
                                                const int*   __restrict__ mask,
                                                const float* __restrict__ bO,
                                                const float* __restrict__ lng,
                                                const float* __restrict__ lnb,
                                                const unsigned short* __restrict__ wsu,
                                                float* __restrict__ out) {
    __shared__ __align__(16) unsigned char smem[SM_TOTAL];
    const __bf16* wsb = (const __bf16*)wsu;
    const int tid  = threadIdx.x;
    const int wave = tid >> 5;
    const int lane = tid & 31;
    const int b0   = blockIdx.x * TB;

    __bf16* qA   = (__bf16*)(smem + SM_QA);   // [16][288]
    __bf16* tq   = (__bf16*)(smem + SM_TQ);   // [2][16][160]
    __bf16* tmat = (__bf16*)(smem + SM_T);    // [2][16][448]
    __bf16* cA   = (__bf16*)(smem + SM_CA);   // [2][16][448]
    float*  s_sm = (float*)(smem + SM_SC);    // [2][20]
    float*  aw   = s_sm + 40;                 // [2][20]

    // ---- S1: build q tile (bf16) and zero TqA padding ----
    for (int idx = tid; idx < TB * OUT_P; idx += 256) {
        int r = idx / OUT_P, c = idx % OUT_P;
        float v = 0.f;
        if (c < NODE_D)      v = node[(size_t)(b0 + r) * NODE_D + c];
        else if (c < OUT_D)  v = timef[(size_t)(b0 + r) * TIME_D + (c - NODE_D)];
        qA[idx] = (__bf16)v;
    }
    for (int idx = tid; idx < 2 * TB * HDP; idx += 256) tq[idx] = (__bf16)0.f;
    __syncthreads();

    // ---- S2: Q = q_in @ W_Q^T  (16x272, K=272) -> per-head A tiles in LDS ----
    {
        const __bf16* wqf = wsb + OFF_WQF;
        for (int nt = wave; nt < QF_NT; nt += 8) {
            v8f acc = {};
            for (int kt = 0; kt < QF_KT; kt++) {
                if (kt + 1 < QF_KT)
                    __builtin_prefetch(wqf + ((kt + 1) * QF_NT + nt) * 512 + lane * 16, 0, 1);
                v16bf a  = load_a_frag(qA, OUT_P, kt, lane);
                v16bf bb = load_b_frag(wqf + (kt * QF_NT + nt) * 512, lane);
                acc = wmma_bf16(a, bb, acc);
            }
            int col  = nt * 16 + (lane & 15);     // < 272
            int head = col / HD;
            int d    = col - head * HD;
#pragma unroll
            for (int g = 0; g < 8; g++) {
                int row = ((lane & 16) ? 8 : 0) + g;
                tq[(head * TB + row) * HDP + d] = (__bf16)acc[g];
            }
        }
    }
    __syncthreads();

    // ---- S3: T[b,h,:] = Q_h[b,:] @ W_K[h-block]  (16x444 per head, K=136) ----
    {
        const __bf16* wkf = wsb + OFF_WKF;
        for (int job = wave; job < 2 * KF_NT; job += 8) {
            int h = job / KF_NT, nt = job % KF_NT;
            v8f acc = {};
            const __bf16* abase = tq + h * TB * HDP;
            for (int kt = 0; kt < KF_KT; kt++) {
                v16bf a  = load_a_frag(abase, HDP, kt, lane);
                v16bf bb = load_b_frag(wkf + (h * KF_KT * KF_NT + kt * KF_NT + nt) * 512, lane);
                acc = wmma_bf16(a, bb, acc);
            }
            int col = nt * 16 + (lane & 15);      // < 448
#pragma unroll
            for (int g = 0; g < 8; g++) {
                int row = ((lane & 16) ? 8 : 0) + g;
                tmat[(h * TB + row) * KEY_P + col] = (__bf16)acc[g];
            }
        }
    }
    __syncthreads();

    // ---- S4: per-element attention core, kv streamed via TDM double buffer ----
#if HAVE_TDM
    if (wave == 0) issue_kv(nbrN, nbrE, nbrT, b0, 0);
#endif
    for (int b = 0; b < TB; b++) {
#if HAVE_TDM
        if (wave == 0) {
            if (b + 1 < TB) {
                issue_kv(nbrN, nbrE, nbrT, b0 + b + 1, (b + 1) & 1);
                __builtin_amdgcn_s_wait_tensorcnt(3);   // current element's 3 loads done
            } else {
                __builtin_amdgcn_s_wait_tensorcnt(0);
            }
        }
#else
        {   // fallback: cooperative synchronous load of kv tile
            const float* srcN = nbrN + (size_t)(b0 + b) * NBR * NODE_D;
            const float* srcE = nbrE + (size_t)(b0 + b) * NBR * EDGE_D;
            const float* srcT = nbrT + (size_t)(b0 + b) * NBR * TIME_D;
            float* dst = (float*)(smem + SM_KV + (b & 1) * KV_BYTES);
            for (int idx = tid; idx < KV_FLOATS; idx += 256) {
                float v;
                if (idx < NBR * NODE_D)          v = srcN[idx];
                else if (idx < 2 * NBR * NODE_D) v = srcE[idx - NBR * NODE_D];
                else                             v = srcT[idx - 2 * NBR * NODE_D];
                dst[idx] = v;
            }
        }
#endif
        __syncthreads();

        const float* kvA = (const float*)(smem + SM_KV + (b & 1) * KV_BYTES);
        const float* kvB = kvA + NBR * NODE_D;
        const float* kvC = kvA + 2 * NBR * NODE_D;

        // scores[h][n] = kv[n,:] . T[b,h,:]   (40 pairs x 4-way k-split)
        if (tid < 160) {
            int p = tid >> 2, sub = tid & 3;
            int h = p / NBR, n = p % NBR;
            const __bf16* Th = tmat + (h * TB + b) * KEY_P;
            float acc = 0.f;
            for (int k = sub; k < NODE_D; k += 4) acc += kvA[n * NODE_D + k] * (float)Th[k];
            for (int k = sub; k < EDGE_D; k += 4) acc += kvB[n * EDGE_D + k] * (float)Th[NODE_D + k];
            for (int k = sub; k < TIME_D; k += 4) acc += kvC[n * TIME_D + k] * (float)Th[NODE_D + EDGE_D + k];
            acc += __shfl_xor(acc, 1);
            acc += __shfl_xor(acc, 2);
            if (sub == 0) s_sm[h * NBR + n] = acc;
        }
        __syncthreads();

        // masked softmax per head
        if (tid < 2) {
            int h = tid;
            const int* mrow = mask + (size_t)(b0 + b) * NBR;
            float sc[NBR];
            float mx = -3.0e38f;
#pragma unroll
            for (int n = 0; n < NBR; n++) {
                float v = s_sm[h * NBR + n] * INV_SQRT_HD;
                if (mrow[n] == 0) v = -1e10f;
                sc[n] = v;
                mx = fmaxf(mx, v);
            }
            float sum = 0.f;
#pragma unroll
            for (int n = 0; n < NBR; n++) { float e = expf(sc[n] - mx); aw[h * NBR + n] = e; sum += e; }
            float inv = 1.f / sum;
#pragma unroll
            for (int n = 0; n < NBR; n++) aw[h * NBR + n] *= inv;
        }
        __syncthreads();

        // c[h][k] = sum_n aw[h][n] * kv[n][k]  -> row b of per-head A tile
        for (int idx = tid; idx < 2 * KEY_P; idx += 256) {
            int h = idx / KEY_P, k = idx % KEY_P;
            float acc = 0.f;
            if (k < NODE_D) {
#pragma unroll
                for (int n = 0; n < NBR; n++) acc += aw[h * NBR + n] * kvA[n * NODE_D + k];
            } else if (k < NODE_D + EDGE_D) {
                int kk = k - NODE_D;
#pragma unroll
                for (int n = 0; n < NBR; n++) acc += aw[h * NBR + n] * kvB[n * EDGE_D + kk];
            } else if (k < KEY_D) {
                int kk = k - NODE_D - EDGE_D;
#pragma unroll
                for (int n = 0; n < NBR; n++) acc += aw[h * NBR + n] * kvC[n * TIME_D + kk];
            }
            cA[(h * TB + b) * KEY_P + k] = (__bf16)acc;
        }
        __syncthreads();   // all reads of this kv buffer done before TDM overwrites it
    }

    // ---- S5: O[b, h*136+d] = c[b,h,:] @ W_V[h]^T  (K=444 per head) ----
    __bf16* oA = (__bf16*)(smem + SM_KV);     // reuse kv region: [16][288] bf16
    for (int idx = tid; idx < TB * OUT_P; idx += 256) oA[idx] = (__bf16)0.f;
    __syncthreads();
    {
        const __bf16* wvf = wsb + OFF_WVF;
        for (int job = wave; job < 2 * VF_NT; job += 8) {
            int h = job / VF_NT, nt = job % VF_NT;
            v8f acc = {};
            const __bf16* abase = cA + h * TB * KEY_P;
            for (int kt = 0; kt < VF_KT; kt++) {
                if (kt + 1 < VF_KT)
                    __builtin_prefetch(wvf + (h * VF_KT * VF_NT + (kt + 1) * VF_NT + nt) * 512 + lane * 16, 0, 1);
                v16bf a  = load_a_frag(abase, KEY_P, kt, lane);
                v16bf bb = load_b_frag(wvf + (h * VF_KT * VF_NT + kt * VF_NT + nt) * 512, lane);
                acc = wmma_bf16(a, bb, acc);
            }
            int lc = nt * 16 + (lane & 15);
            if (lc < HD) {
                int col = h * HD + lc;
#pragma unroll
                for (int g = 0; g < 8; g++) {
                    int row = ((lane & 16) ? 8 : 0) + g;
                    oA[row * OUT_P + col] = (__bf16)acc[g];
                }
            }
        }
    }
    __syncthreads();

    // ---- S6: out = O @ W_O^T + b_O + residual -> xbuf ----
    float* xbuf = (float*)(smem + SM_T);      // reuse T region: [16][272] f32
    {
        const __bf16* wof = wsb + OFF_WOF;
        for (int nt = wave; nt < OF_NT; nt += 8) {
            v8f acc = {};
            for (int kt = 0; kt < OF_KT; kt++) {
                v16bf a  = load_a_frag(oA, OUT_P, kt, lane);
                v16bf bb = load_b_frag(wof + (kt * OF_NT + nt) * 512, lane);
                acc = wmma_bf16(a, bb, acc);
            }
            int col = nt * 16 + (lane & 15);  // < 272
            float bias = bO[col];
#pragma unroll
            for (int g = 0; g < 8; g++) {
                int row = ((lane & 16) ? 8 : 0) + g;
                float resid = (col < NODE_D)
                                  ? node[(size_t)(b0 + row) * NODE_D + col]
                                  : timef[(size_t)(b0 + row) * TIME_D + (col - NODE_D)];
                xbuf[row * OUT_D + col] = acc[g] + bias + resid;
            }
        }
    }
    __syncthreads();

    // ---- LayerNorm + store (16 lanes per row, shfl reductions) ----
    {
        int row = tid >> 4, sub = tid & 15;
        const float* xr = xbuf + row * OUT_D;
        float s = 0.f;
        for (int j = sub; j < OUT_D; j += 16) s += xr[j];
        s += __shfl_xor(s, 1); s += __shfl_xor(s, 2); s += __shfl_xor(s, 4); s += __shfl_xor(s, 8);
        float mu = s * (1.f / OUT_D);
        float v = 0.f;
        for (int j = sub; j < OUT_D; j += 16) { float d = xr[j] - mu; v += d * d; }
        v += __shfl_xor(v, 1); v += __shfl_xor(v, 2); v += __shfl_xor(v, 4); v += __shfl_xor(v, 8);
        float rstd = rsqrtf(v * (1.f / OUT_D) + LN_EPS);
        float* orow = out + (size_t)(b0 + row) * OUT_D;
        for (int j = sub; j < OUT_D; j += 16)
            orow[j] = (xr[j] - mu) * rstd * lng[j] + lnb[j];
    }
}

// ---------------- host entry --------------------------------------------------
extern "C" void kernel_launch(void* const* d_in, const int* in_sizes, int n_in,
                              void* d_out, int out_size, void* d_ws, size_t ws_size,
                              hipStream_t stream) {
    (void)in_sizes; (void)n_in; (void)out_size; (void)ws_size;
    const float* node  = (const float*)d_in[0];
    const float* timef = (const float*)d_in[1];
    const float* nbrN  = (const float*)d_in[2];
    const float* nbrT  = (const float*)d_in[3];
    const float* nbrE  = (const float*)d_in[4];
    const int*   mask  = (const int*)d_in[5];
    const float* WQ    = (const float*)d_in[6];
    const float* WK    = (const float*)d_in[7];
    const float* WV    = (const float*)d_in[8];
    const float* WO    = (const float*)d_in[9];
    const float* bO    = (const float*)d_in[10];
    const float* lng   = (const float*)d_in[11];
    const float* lnb   = (const float*)d_in[12];
    unsigned short* ws = (unsigned short*)d_ws;

    tga_prep<<<WS_HALVES / 256, 256, 0, stream>>>(WQ, WK, WV, WO, ws);
    tga_main<<<BATCH / TB, 256, 0, stream>>>(node, timef, nbrN, nbrT, nbrE, mask,
                                             bO, lng, lnb, ws, (float*)d_out);
}
